// FeatupNei_48326972014844
// MI455X (gfx1250) — matrix operand order repaired and checked by reading, not compiled, not run on hardware
//
#include <hip/hip_runtime.h>
#include <math.h>

// ---------------- problem constants ----------------
#define BB   4
#define NN   80
#define CC   384
#define DD   386      // C + 2
#define DP   416      // DD padded to multiple of 32 (WMMA K granularity)
#define DFFC 2048
#define T3   240      // 3*N decoder tokens
#define T9   720      // 9*N memory tokens
#define T9P  736      // T9 padded to 32
#define T3P  256      // T3 padded to 32
#define RPAD 768      // row padding (mult of 64) for N-major B sources
#define VROWS 448     // feature rows of transposed V (DD padded to 64)
#define HH   224

typedef __attribute__((ext_vector_type(16))) _Float16 v16h;
typedef __attribute__((ext_vector_type(8)))  _Float16 h8;
typedef __attribute__((ext_vector_type(8)))  float    v8f;

union Frag16 { v16h v; unsigned u[8]; };

__device__ __forceinline__ void async_ld16(unsigned lds, unsigned long long gp) {
  asm volatile("global_load_async_to_lds_b128 %0, %1, off" :: "v"(lds), "v"(gp) : "memory");
}
__device__ __forceinline__ void async_ld32(unsigned lds, unsigned long long gp) {
  // INST_OFFSET applies to both LDS and global sides (ISA 08_async_tensor 4.4)
  asm volatile("global_load_async_to_lds_b128 %0, %1, off" :: "v"(lds), "v"(gp) : "memory");
  asm volatile("global_load_async_to_lds_b128 %0, %1, off offset:16" :: "v"(lds), "v"(gp) : "memory");
}

// =====================================================================
// WMMA GEMM:  O = A(MxK,f16,row-major) * B(f16, stored N-major: N rows x
// ldb>=K cols) + bias, optional relu, optional transposed f16 store.
// Double-buffered CDNA5 async global->LDS staging (ASYNCcnt): tile i+1
// streams into the alternate LDS buffer while tile i feeds the WMMAs;
// s_wait_asynccnt 0x3 retires exactly the previous tile's 3 async ops.
// Tile: 128x64x32, 8 waves, 32x32 per wave. K must be a multiple of 32.
// =====================================================================
__global__ __launch_bounds__(256) void k_gemm(
    const _Float16* __restrict__ A, const _Float16* __restrict__ Bm,
    const float* __restrict__ bias, float* __restrict__ O32,
    _Float16* __restrict__ O16, int M, int N, int K,
    int lda, int ldb, int ldo, int ldo16,
    long sA, long sB, long sO, long sO16, int flags) {
  __shared__ _Float16 As[2][128][40];   // padded rows (80B stride, 16B aligned)
  __shared__ _Float16 Bs[2][64][40];    // N-major: Bs[.][n][k]
  const int bz = blockIdx.z;
  const _Float16* gA = A + (long)bz * sA;
  const _Float16* gB = Bm + (long)bz * sB;
  const int m0 = blockIdx.x * 128;
  const int n0 = blockIdx.y * 64;
  const int t = threadIdx.x;
  const int lane = t & 31;
  const int wid = t >> 5;
  const int wm = (wid & 3) * 32;   // wave M offset in tile
  const int wn = (wid >> 2) * 32;  // wave N offset in tile

  v8f acc[2][2];
#pragma unroll
  for (int mi = 0; mi < 2; ++mi)
#pragma unroll
    for (int ni = 0; ni < 2; ++ni)
#pragma unroll
      for (int r = 0; r < 8; ++r) acc[mi][ni][r] = 0.f;

  // loader mappings (all 256 threads active, no divergence at async ops)
  const int arow = t >> 1, ahalf = (t & 1) * 16;  // A: 128 rows x 32 k
  const int brow = t >> 2, bk8 = (t & 3) * 8;     // B: 64 rows x 32 k
  const int arowc = (m0 + arow < M) ? (m0 + arow) : (M - 1);  // clamp (rows >= M masked at store)

  const unsigned ldsA0 = (unsigned)(unsigned long long)&As[0][arow][ahalf];
  const unsigned ldsA1 = (unsigned)(unsigned long long)&As[1][arow][ahalf];
  const unsigned ldsB0 = (unsigned)(unsigned long long)&Bs[0][brow][bk8];
  const unsigned ldsB1 = (unsigned)(unsigned long long)&Bs[1][brow][bk8];
  const _Float16* pA = gA + (long)arowc * lda + ahalf;
  const _Float16* pB = gB + (long)(n0 + brow) * ldb + bk8;

  const int nt = K >> 5;  // K is always a multiple of 32 here
  // prologue: stage tile 0 into buffer 0
  async_ld32(ldsA0, (unsigned long long)pA);
  async_ld16(ldsB0, (unsigned long long)pB);

  for (int i = 0; i < nt; ++i) {
    const int buf = i & 1;
    if (i + 1 < nt) {
      const int nb = buf ^ 1;
      async_ld32(nb ? ldsA1 : ldsA0, (unsigned long long)(pA + (long)(i + 1) * 32));
      async_ld16(nb ? ldsB1 : ldsB0, (unsigned long long)(pB + (long)(i + 1) * 32));
      asm volatile("s_wait_asynccnt 0x3" ::: "memory");  // tile i done, i+1 in flight
    } else {
      asm volatile("s_wait_asynccnt 0x0" ::: "memory");
    }
    __syncthreads();

    // ---- fragments per ISA VGPR layouts (both sides 2x ds_load_b128) ----
    Frag16 fa[2], fb[2];
    const int ar  = lane & 15;
    const int khi = (lane >> 4) * 8;     // A: lanes16-31 hold K+8 / K+24
#pragma unroll
    for (int mi = 0; mi < 2; ++mi) {
#pragma unroll
      for (int j = 0; j < 8; ++j) {
        int kk = ((j < 4) ? (2 * j) : (16 + 2 * (j - 4))) + khi;
        fa[mi].u[j] = *(const unsigned*)&As[buf][wm + mi * 16 + ar][kk];
      }
    }
    const int kb0 = (lane >> 4) * 16;    // B: lanes16-31 hold K+8 / K+24
#pragma unroll
    for (int ni = 0; ni < 2; ++ni) {
      int col = wn + ni * 16 + (lane & 15);
#pragma unroll
      for (int j = 0; j < 8; ++j)
        fb[ni].u[j] = *(const unsigned*)&Bs[buf][col][((j < 4) ? (2 * j) : (16 + 2 * (j - 4))) + (kb0 >> 1)];
    }
#pragma unroll
    for (int mi = 0; mi < 2; ++mi)
#pragma unroll
      for (int ni = 0; ni < 2; ++ni)
        acc[mi][ni] = __builtin_amdgcn_wmma_f32_16x16x32_f16(
            false, fa[mi].v, false, fb[ni].v, (short)0, acc[mi][ni], false, false);
    __syncthreads();  // protects buffer reuse two iterations ahead
  }

  // ---- store C (VGPR r -> row r + 8*(lane>=16), col = lane&15) ----
#pragma unroll
  for (int mi = 0; mi < 2; ++mi)
#pragma unroll
    for (int ni = 0; ni < 2; ++ni) {
      int col = n0 + wn + ni * 16 + (lane & 15);
      if (col >= N) continue;
      float bv = bias ? bias[col] : 0.f;
#pragma unroll
      for (int r = 0; r < 8; ++r) {
        int row = m0 + wm + mi * 16 + r + ((lane >> 4) * 8);
        if (row >= M) continue;
        float v = acc[mi][ni][r] + bv;
        if (flags & 1) v = fmaxf(v, 0.f);
        if (O32) O32[(long)bz * sO + (long)row * ldo + col] = v;
        if (O16) {
          long i16 = (flags & 4) ? ((long)col * ldo16 + row) : ((long)row * ldo16 + col);
          O16[(long)bz * sO16 + i16] = (_Float16)v;
        }
      }
    }
}

// ---------------- softmax over rows (scaled), f32 -> padded f16 -------
__global__ __launch_bounds__(256) void k_softmax(
    const float* __restrict__ S, _Float16* __restrict__ P,
    int nk, int lds_, int ldp, long sS, long sP, float scale) {
  __shared__ float red[256];
  const int b = blockIdx.y, row = blockIdx.x, tid = threadIdx.x;
  const float* sr = S + (long)b * sS + (long)row * lds_;
  _Float16* pr = P + (long)b * sP + (long)row * ldp;
  float ev[3];
  float m = -3.0e38f;
  for (int i = tid; i < nk; i += 256) m = fmaxf(m, sr[i] * scale);
  red[tid] = m; __syncthreads();
  for (int st = 128; st > 0; st >>= 1) { if (tid < st) red[tid] = fmaxf(red[tid], red[tid + st]); __syncthreads(); }
  float gm = red[0]; __syncthreads();
  float sum = 0.f; int c = 0;
  for (int i = tid; i < nk; i += 256, ++c) { float e = __expf(sr[i] * scale - gm); ev[c] = e; sum += e; }
  red[tid] = sum; __syncthreads();
  for (int st = 128; st > 0; st >>= 1) { if (tid < st) red[tid] += red[tid + st]; __syncthreads(); }
  float inv = 1.f / red[0];
  c = 0;
  for (int i = tid; i < nk; i += 256, ++c) pr[i] = (_Float16)(ev[c] * inv);
}

// ---------------- layernorm (+optional residual, +optional PE) --------
__global__ __launch_bounds__(128) void k_ln(
    const float* __restrict__ X, const float* __restrict__ R,
    const float* __restrict__ g, const float* __restrict__ bt,
    const float* __restrict__ pe, float* __restrict__ O32,
    _Float16* __restrict__ O16, int d,
    long sX, long sR, long sO, long sO16, int ldo16) {
  __shared__ float red[128];
  const int b = blockIdx.y, row = blockIdx.x, tid = threadIdx.x;
  const float* xr = X + (long)b * sX + (long)row * d;
  const float* rr = R ? (R + (long)b * sR + (long)row * d) : (const float*)0;
  float v[4]; int c = 0; float sum = 0.f, sq = 0.f;
  for (int i = tid; i < d; i += 128, ++c) {
    float t = xr[i]; if (rr) t += rr[i];
    v[c] = t; sum += t; sq += t * t;
  }
  red[tid] = sum; __syncthreads();
  for (int st = 64; st > 0; st >>= 1) { if (tid < st) red[tid] += red[tid + st]; __syncthreads(); }
  float mean = red[0] / d; __syncthreads();
  red[tid] = sq; __syncthreads();
  for (int st = 64; st > 0; st >>= 1) { if (tid < st) red[tid] += red[tid + st]; __syncthreads(); }
  float var = red[0] / d - mean * mean;
  float rs = rsqrtf(var + 1e-5f);
  c = 0;
  for (int i = tid; i < d; i += 128, ++c) {
    float y = (v[c] - mean) * rs * g[i] + bt[i];
    if (pe) y += pe[(long)row * d + i];
    if (O32) O32[(long)b * sO + (long)row * d + i] = y;
    if (O16) O16[(long)b * sO16 + (long)row * ldo16 + i] = (_Float16)y;
  }
}

// ---- weight convert: W(OxI)f32 -> Wt f16 stored N-major (NP rows x KP)
__global__ void k_wconv(const float* __restrict__ W, _Float16* __restrict__ Wt,
                        int I, int O, int KP, int NP) {
  long idx = (long)blockIdx.x * 256 + threadIdx.x;
  if (idx >= (long)KP * NP) return;
  int k = (int)(idx % KP); int n = (int)(idx / KP);
  float v = (k < I && n < O) ? W[(long)n * I + k] : 0.f;
  Wt[idx] = (_Float16)v;
}

// ---------------- 2x2 avg pool (square maps) --------------------------
__global__ void k_pool2(const float* __restrict__ in, float* __restrict__ out, int H) {
  int H2 = H >> 1;
  long idx = (long)blockIdx.x * 256 + threadIdx.x;
  long total = (long)BB * CC * H2 * H2;
  if (idx >= total) return;
  int x = (int)(idx % H2); long r = idx / H2; int y = (int)(r % H2); long bc = r / H2;
  const float* p = in + (bc * H + 2 * y) * H + 2 * x;
  out[idx] = 0.25f * (p[0] + p[1] + p[H] + p[H + 1]);
}

// ---------------- boundary shift alignment ----------------------------
__global__ __launch_bounds__(128) void k_shift(const int* __restrict__ fir,
                                               const int* __restrict__ pre,
                                               int* __restrict__ out) {
  __shared__ int f[NN * 2], p[NN * 2], dist[NN];
  __shared__ int best;
  int b = blockIdx.x, tid = threadIdx.x;
  for (int i = tid; i < NN * 2; i += 128) { f[i] = fir[(long)b * NN * 2 + i]; p[i] = pre[(long)b * NN * 2 + i]; }
  __syncthreads();
  if (tid < NN) {
    int s = 0;
    for (int j = 0; j < NN; ++j) {
      int src = j - tid; src += (src < 0) ? NN : 0;
      int d0 = p[2 * j] - f[2 * src]; int d1 = p[2 * j + 1] - f[2 * src + 1];
      s += (d0 < 0 ? -d0 : d0) + (d1 < 0 ? -d1 : d1);
    }
    dist[tid] = s;
  }
  __syncthreads();
  if (tid == 0) { int bi = 0; for (int i = 1; i < NN; ++i) if (dist[i] < dist[bi]) bi = i; best = bi; }
  __syncthreads();
  if (tid < NN) {
    int src = tid - best; src += (src < 0) ? NN : 0;
    out[((long)b * NN + tid) * 2 + 0] = f[2 * src];
    out[((long)b * NN + tid) * 2 + 1] = f[2 * src + 1];
  }
}

// ---------------- token gathers ---------------------------------------
__global__ void k_gather_int(const float* __restrict__ F, const int* __restrict__ bou,
                             float* __restrict__ xraw, int rowoff) {
  int b = blockIdx.y, j = blockIdx.x;
  int y = bou[((long)b * NN + j) * 2 + 0];
  int x = bou[((long)b * NN + j) * 2 + 1];
  float* dst = xraw + ((long)b * T3 + rowoff + j) * DD;
  for (int c = threadIdx.x; c < CC; c += blockDim.x)
    dst[c] = F[(((long)b * CC + c) * HH + y) * HH + x];
  if (threadIdx.x == 0) { dst[CC] = (float)y; dst[CC + 1] = (float)x; }
}

__global__ void k_gather_cur(const float* __restrict__ F, const float* __restrict__ curb,
                             float* __restrict__ xraw, int* __restrict__ cbi) {
  int b = blockIdx.y, j = blockIdx.x;
  float yf = fminf(fmaxf(curb[((long)b * NN + j) * 2 + 0], 0.f), 223.f);
  float xf = fminf(fmaxf(curb[((long)b * NN + j) * 2 + 1], 0.f), 223.f);
  int yi = (int)yf, xi = (int)xf;
  float* dst = xraw + ((long)b * T3 + NN + j) * DD;
  for (int c = threadIdx.x; c < CC; c += blockDim.x)
    dst[c] = F[(((long)b * CC + c) * HH + yi) * HH + xi];
  if (threadIdx.x == 0) {
    cbi[((long)b * NN + j) * 2 + 0] = yi;
    cbi[((long)b * NN + j) * 2 + 1] = xi;
    dst[CC] = yf; dst[CC + 1] = xf;
  }
}

// neighbor tokens at scale s (reference indexes padded map at bou//k + d,
// i.e. original offset d-1), emitted directly as f16 GEMM inputs.
__global__ void k_neighbor(const float* __restrict__ Pm, const int* __restrict__ cbi,
                           _Float16* __restrict__ NT, int Hs, int sft) {
  int b = blockIdx.y, t = blockIdx.x;
  int j = t % NN, d9 = t / NN;
  int dx = d9 / 3 - 1, dy = d9 % 3 - 1;
  int y = (cbi[((long)b * NN + j) * 2 + 0] >> sft) + dx - 1;
  int x = (cbi[((long)b * NN + j) * 2 + 1] >> sft) + dy - 1;
  bool ok = (y >= 0) && (y < Hs) && (x >= 0) && (x < Hs);
  _Float16* dst = NT + ((long)b * T9 + t) * CC;
  for (int c = threadIdx.x; c < CC; c += blockDim.x) {
    float v = ok ? Pm[(((long)b * CC + c) * Hs + y) * Hs + x] : 0.f;
    dst[c] = (_Float16)v;
  }
}

// ---------------- misc small kernels ----------------------------------
__global__ void k_i2f(const int* __restrict__ in, float* __restrict__ out, int n) {
  int i = blockIdx.x * 256 + threadIdx.x;
  if (i < n) out[i] = (float)in[i];
}

__global__ void k_pe(float* __restrict__ pe, int rows) {
  long idx = (long)blockIdx.x * 256 + threadIdx.x;
  if (idx >= (long)rows * DD) return;
  int p = (int)(idx / DD), d = (int)(idx % DD);
  int i2 = (d >> 1) << 1;
  float div = __expf((float)i2 * (-logf(10000.f) / (float)DD));
  float a = (float)p * div;
  pe[idx] = (d & 1) ? cosf(a) : sinf(a);
}

__global__ void k_offset(const float* __restrict__ xw, const float* __restrict__ ow,
                         const float* __restrict__ ob, const int* __restrict__ preb,
                         float* __restrict__ outp, float* __restrict__ curb) {
  int idx = blockIdx.x * blockDim.x + threadIdx.x;
  if (idx >= BB * NN) return;
  int b = idx / NN, j = idx % NN;
  const float* row = xw + ((long)b * T3 + NN + j) * DD;
  float o0 = ob[0], o1 = ob[1];
  for (int d = 0; d < DD; ++d) { float t = row[d]; o0 += t * ow[d]; o1 += t * ow[DD + d]; }
  float r0 = o0 + (float)preb[idx * 2], r1 = o1 + (float)preb[idx * 2 + 1];
  outp[idx * 2] = r0; outp[idx * 2 + 1] = r1;
  curb[idx * 2] = r0; curb[idx * 2 + 1] = r1;
}

// =====================================================================
extern "C" void kernel_launch(void* const* d_in, const int* in_sizes, int n_in,
                              void* d_out, int out_size, void* d_ws, size_t ws_size,
                              hipStream_t stream) {
  (void)in_sizes; (void)n_in; (void)out_size; (void)ws_size;
  const float* fir_feats = (const float*)d_in[0];
  const int*   fir_bou   = (const int*)d_in[1];
  const float* pre_feats = (const float*)d_in[2];
  const float* cur_feats = (const float*)d_in[3];
  const int*   pre_bou   = (const int*)d_in[4];
  // params flattened (jax tree order: sorted dict keys)
  const float* dec_b1    = (const float*)d_in[5];
  const float* dec_b2    = (const float*)d_in[6];
  const float* dec_ca_bk = (const float*)d_in[7];
  const float* dec_ca_bo = (const float*)d_in[8];
  const float* dec_ca_bq = (const float*)d_in[9];
  const float* dec_ca_bv = (const float*)d_in[10];
  const float* dec_ca_wk = (const float*)d_in[11];
  const float* dec_ca_wo = (const float*)d_in[12];
  const float* dec_ca_wq = (const float*)d_in[13];
  const float* dec_ca_wv = (const float*)d_in[14];
  const float* dec_ln1b  = (const float*)d_in[15];
  const float* dec_ln1g  = (const float*)d_in[16];
  const float* dec_ln2b  = (const float*)d_in[17];
  const float* dec_ln2g  = (const float*)d_in[18];
  const float* dec_ln3b  = (const float*)d_in[19];
  const float* dec_ln3g  = (const float*)d_in[20];
  const float* dec_sa_bk = (const float*)d_in[21];
  const float* dec_sa_bo = (const float*)d_in[22];
  const float* dec_sa_bq = (const float*)d_in[23];
  const float* dec_sa_bv = (const float*)d_in[24];
  const float* dec_sa_wk = (const float*)d_in[25];
  const float* dec_sa_wo = (const float*)d_in[26];
  const float* dec_sa_wq = (const float*)d_in[27];
  const float* dec_sa_wv = (const float*)d_in[28];
  const float* dec_w1    = (const float*)d_in[29];
  const float* dec_w2    = (const float*)d_in[30];
  const float* enc_b1    = (const float*)d_in[31];
  const float* enc_b2    = (const float*)d_in[32];
  const float* enc_ln1b  = (const float*)d_in[33];
  const float* enc_ln1g  = (const float*)d_in[34];
  const float* enc_ln2b  = (const float*)d_in[35];
  const float* enc_ln2g  = (const float*)d_in[36];
  const float* enc_sa_bk = (const float*)d_in[37];
  const float* enc_sa_bo = (const float*)d_in[38];
  const float* enc_sa_bq = (const float*)d_in[39];
  const float* enc_sa_bv = (const float*)d_in[40];
  const float* enc_sa_wk = (const float*)d_in[41];
  const float* enc_sa_wo = (const float*)d_in[42];
  const float* enc_sa_wq = (const float*)d_in[43];
  const float* enc_sa_wv = (const float*)d_in[44];
  const float* enc_w1    = (const float*)d_in[45];
  const float* enc_w2    = (const float*)d_in[46];
  const float* fc_b      = (const float*)d_in[47];
  const float* fc_w      = (const float*)d_in[48];
  const float* ln_b      = (const float*)d_in[49];
  const float* ln_g      = (const float*)d_in[50];
  const float* off_b     = (const float*)d_in[51];
  const float* off_w     = (const float*)d_in[52];

  // ---------------- workspace carve ----------------
  char* w = (char*)d_ws;
  size_t off = 0;
  auto A16 = [&](size_t e) { _Float16* p = (_Float16*)(w + off); off = (off + e * 2 + 255) & ~(size_t)255; return p; };
  auto A32 = [&](size_t e) { float* p = (float*)(w + off); off = (off + e * 4 + 255) & ~(size_t)255; return p; };
  auto AI  = [&](size_t e) { int* p = (int*)(w + off); off = (off + e * 4 + 255) & ~(size_t)255; return p; };

  // f16 arena (zeroed once per launch => padding lanes are guaranteed 0)
  _Float16* x16   = A16((size_t)BB * T3 * DP);
  _Float16* ntok  = A16((size_t)BB * T9 * CC);
  _Float16* t16   = A16((size_t)BB * T9 * DP);
  _Float16* a16   = A16((size_t)BB * T9 * DP);
  _Float16* q16   = A16((size_t)BB * RPAD * DP);
  _Float16* k16   = A16((size_t)BB * RPAD * DP);
  _Float16* v16t  = A16((size_t)BB * VROWS * RPAD);  // V stored N-major (feature, token)
  _Float16* p9    = A16((size_t)BB * T9 * T9P);
  _Float16* p3    = A16((size_t)BB * T3 * T3P);
  _Float16* h16   = A16((size_t)BB * T9 * DFFC);
  _Float16* mem16[4];
  for (int s = 0; s < 4; ++s) mem16[s] = A16((size_t)BB * T9 * DP);
  // converted weights, N-major (NP rows x KP cols); rewritten each launch
  _Float16* fcw_t = A16((size_t)448 * 384);
  _Float16* e_wq  = A16((size_t)448 * DP); _Float16* e_wk = A16((size_t)448 * DP);
  _Float16* e_wv  = A16((size_t)448 * DP); _Float16* e_wo = A16((size_t)448 * DP);
  _Float16* e_w1t = A16((size_t)2048 * DP); _Float16* e_w2t = A16((size_t)448 * 2048);
  _Float16* dsa_wq = A16((size_t)448 * DP); _Float16* dsa_wk = A16((size_t)448 * DP);
  _Float16* dsa_wv = A16((size_t)448 * DP); _Float16* dsa_wo = A16((size_t)448 * DP);
  _Float16* dca_wq = A16((size_t)448 * DP); _Float16* dca_wk = A16((size_t)448 * DP);
  _Float16* dca_wv = A16((size_t)448 * DP); _Float16* dca_wo = A16((size_t)448 * DP);
  _Float16* d_w1t = A16((size_t)2048 * DP); _Float16* d_w2t = A16((size_t)448 * 2048);
  size_t f16_end = off;
  // f32 / int buffers
  float* pe     = A32((size_t)T9 * DD);
  float* pool1  = A32((size_t)BB * CC * 112 * 112);
  float* pool2  = A32((size_t)BB * CC * 56 * 56);
  float* pool3  = A32((size_t)BB * CC * 28 * 28);
  float* x_raw  = A32((size_t)BB * T3 * DD);
  float* xw     = A32((size_t)BB * T3 * DD);
  float* t_f32  = A32((size_t)BB * T9 * DD);
  float* tmp9   = A32((size_t)BB * T9 * DD);
  float* tmp3   = A32((size_t)BB * T3 * DD);
  float* s_f32  = A32((size_t)BB * T9 * T9);
  float* cur_b  = A32((size_t)BB * NN * 2);
  int*   cb_i   = AI((size_t)BB * NN * 2);
  int*   fir_al = AI((size_t)BB * NN * 2);

  hipMemsetAsync(d_ws, 0, f16_end, stream);

  const float scl = 1.0f / sqrtf((float)DD);

  auto gemm = [&](const _Float16* A, const _Float16* Bm, const float* bias,
                  float* O32, _Float16* O16, int M, int Nn, int K,
                  int lda, int ldb, int ldo, int ldo16,
                  long sA, long sB, long sO, long sO16, int flags) {
    dim3 g((M + 127) / 128, (Nn + 63) / 64, BB);
    k_gemm<<<g, dim3(256), 0, stream>>>(A, Bm, bias, O32, O16, M, Nn, K,
                                        lda, ldb, ldo, ldo16, sA, sB, sO, sO16, flags);
  };
  auto wconv = [&](const float* W, _Float16* Wt, int I, int O) {
    int KP = (I + 31) & ~31; int NP = (O + 63) & ~63;
    long tot = (long)KP * NP;
    k_wconv<<<dim3((unsigned)((tot + 255) / 256)), dim3(256), 0, stream>>>(W, Wt, I, O, KP, NP);
  };
  auto lnk = [&](const float* X, const float* R, const float* g, const float* bt,
                 const float* pe_, float* O32, _Float16* O16, int rows,
                 long sX, long sR, long sO, long sO16) {
    k_ln<<<dim3(rows, BB), 128, 0, stream>>>(X, R, g, bt, pe_, O32, O16, DD,
                                             sX, sR, sO, sO16, DP);
  };
  auto attn = [&](const _Float16* xin, long sXin, int nq,
                  const _Float16* kvin, long sKV, int nk, int nkP,
                  _Float16* wq_, const float* bq_, _Float16* wk_, const float* bk_,
                  _Float16* wv_, const float* bv_, _Float16* wo_, const float* bo_,
                  _Float16* pbuf, long sPb, float* outT, long sOutT, long sScore) {
    gemm(xin,  wq_, bq_, nullptr, q16, nq, DD, DP, DP, DP, 0, DP, sXin, 0, 0, (long)RPAD * DP, 0);
    gemm(kvin, wk_, bk_, nullptr, k16, nk, DD, DP, DP, DP, 0, DP, sKV,  0, 0, (long)RPAD * DP, 0);
    // V projection, f16 output stored transposed -> N-major (feature, token)
    gemm(kvin, wv_, bv_, nullptr, v16t, nk, DD, DP, DP, DP, 0, RPAD, sKV, 0, 0, (long)VROWS * RPAD, 4);
    // scores: Q * K^T  (K-matrix rows are tokens = N-major source)
    gemm(q16, k16, nullptr, s_f32, nullptr, nq, nk, DP, DP, DP, nk, 0,
         (long)RPAD * DP, (long)RPAD * DP, sScore, 0, 0);
    k_softmax<<<dim3(nq, BB), 256, 0, stream>>>(s_f32, pbuf, nk, nk, nkP, sScore, sPb, scl);
    // P * V (v16t is N-major over features, ldb spans tokens)
    gemm(pbuf, v16t, nullptr, nullptr, a16, nq, DD, nkP, nkP, RPAD, 0, DP,
         sPb, (long)VROWS * RPAD, 0, (long)T9 * DP, 0);
    gemm(a16, wo_, bo_, outT, nullptr, nq, DD, DP, DP, DP, DD, 0,
         (long)T9 * DP, 0, sOutT, 0, 0);
  };

  // ---------------- setup ----------------
  long peTot = (long)T9 * DD;
  k_pe<<<dim3((unsigned)((peTot + 255) / 256)), 256, 0, stream>>>(pe, T9);
  wconv(fc_w, fcw_t, 384, 386);
  wconv(enc_sa_wq, e_wq, DD, DD); wconv(enc_sa_wk, e_wk, DD, DD);
  wconv(enc_sa_wv, e_wv, DD, DD); wconv(enc_sa_wo, e_wo, DD, DD);
  wconv(enc_w1, e_w1t, DD, DFFC); wconv(enc_w2, e_w2t, DFFC, DD);
  wconv(dec_sa_wq, dsa_wq, DD, DD); wconv(dec_sa_wk, dsa_wk, DD, DD);
  wconv(dec_sa_wv, dsa_wv, DD, DD); wconv(dec_sa_wo, dsa_wo, DD, DD);
  wconv(dec_ca_wq, dca_wq, DD, DD); wconv(dec_ca_wk, dca_wk, DD, DD);
  wconv(dec_ca_wv, dca_wv, DD, DD); wconv(dec_ca_wo, dca_wo, DD, DD);
  wconv(dec_w1, d_w1t, DD, DFFC); wconv(dec_w2, d_w2t, DFFC, DD);
  {
    long t1 = (long)BB * CC * 112 * 112;
    k_pool2<<<dim3((unsigned)((t1 + 255) / 256)), 256, 0, stream>>>(cur_feats, pool1, 224);
    long t2 = (long)BB * CC * 56 * 56;
    k_pool2<<<dim3((unsigned)((t2 + 255) / 256)), 256, 0, stream>>>(pool1, pool2, 112);
    long t3 = (long)BB * CC * 28 * 28;
    k_pool2<<<dim3((unsigned)((t3 + 255) / 256)), 256, 0, stream>>>(pool2, pool3, 56);
  }
  k_shift<<<BB, 128, 0, stream>>>(fir_bou, pre_bou, fir_al);
  k_gather_int<<<dim3(NN, BB), 128, 0, stream>>>(pre_feats, pre_bou, x_raw, 0);
  k_gather_int<<<dim3(NN, BB), 128, 0, stream>>>(fir_feats, fir_al, x_raw, 2 * NN);
  k_i2f<<<dim3((BB * NN * 2 + 255) / 256), 256, 0, stream>>>(pre_bou, cur_b, BB * NN * 2);

  // ---------------- refinement loop ----------------
  for (int r = 0; r < 3; ++r) {
    k_gather_cur<<<dim3(NN, BB), 128, 0, stream>>>(cur_feats, cur_b, x_raw, cb_i);
    lnk(x_raw, nullptr, ln_g, ln_b, pe, xw, x16, T3,
        (long)T3 * DD, 0, (long)T3 * DD, (long)T3 * DP);

    // memories: encoder over 720 neighbor tokens at each scale
    for (int s = 0; s < 4; ++s) {
      const float* src = (s == 0) ? cur_feats : (s == 1) ? pool1 : (s == 2) ? pool2 : pool3;
      int Hs = HH >> s;
      k_neighbor<<<dim3(T9, BB), 128, 0, stream>>>(src, cb_i, ntok, Hs, s);
      gemm(ntok, fcw_t, fc_b, t_f32, nullptr, T9, DD, CC, CC, 384, DD, 0,
           (long)T9 * CC, 0, (long)T9 * DD, 0, 0);
      lnk(t_f32, nullptr, ln_g, ln_b, pe, t_f32, t16, T9,
          (long)T9 * DD, 0, (long)T9 * DD, (long)T9 * DP);
      attn(t16, (long)T9 * DP, T9, t16, (long)T9 * DP, T9, T9P,
           e_wq, enc_sa_bq, e_wk, enc_sa_bk, e_wv, enc_sa_bv, e_wo, enc_sa_bo,
           p9, (long)T9 * T9P, tmp9, (long)T9 * DD, (long)T9 * T9);
      lnk(t_f32, tmp9, enc_ln1g, enc_ln1b, nullptr, t_f32, t16, T9,
          (long)T9 * DD, (long)T9 * DD, (long)T9 * DD, (long)T9 * DP);
      gemm(t16, e_w1t, enc_b1, nullptr, h16, T9, DFFC, DP, DP, DP, 0, DFFC,
           (long)T9 * DP, 0, 0, (long)T9 * DFFC, 1);
      gemm(h16, e_w2t, enc_b2, tmp9, nullptr, T9, DD, DFFC, DFFC, 2048, DD, 0,
           (long)T9 * DFFC, 0, (long)T9 * DD, 0, 0);
      lnk(t_f32, tmp9, enc_ln2g, enc_ln2b, nullptr, nullptr, mem16[s], T9,
          (long)T9 * DD, (long)T9 * DD, 0, (long)T9 * DP);
    }

    // decoder stack (coarse -> fine)
    for (int s = 3; s >= 0; --s) {
      attn(x16, (long)T3 * DP, T3, x16, (long)T3 * DP, T3, T3P,
           dsa_wq, dec_sa_bq, dsa_wk, dec_sa_bk, dsa_wv, dec_sa_bv, dsa_wo, dec_sa_bo,
           p3, (long)T3 * T3P, tmp3, (long)T3 * DD, (long)T3 * T3);
      lnk(xw, tmp3, dec_ln1g, dec_ln1b, nullptr, xw, x16, T3,
          (long)T3 * DD, (long)T3 * DD, (long)T3 * DD, (long)T3 * DP);
      attn(x16, (long)T3 * DP, T3, mem16[s], (long)T9 * DP, T9, T9P,
           dca_wq, dec_ca_bq, dca_wk, dec_ca_bk, dca_wv, dec_ca_bv, dca_wo, dec_ca_bo,
           p9, (long)T9 * T9P, tmp3, (long)T3 * DD, (long)T3 * T9);
      lnk(xw, tmp3, dec_ln2g, dec_ln2b, nullptr, xw, x16, T3,
          (long)T3 * DD, (long)T3 * DD, (long)T3 * DD, (long)T3 * DP);
      gemm(x16, d_w1t, dec_b1, nullptr, h16, T3, DFFC, DP, DP, DP, 0, DFFC,
           (long)T3 * DP, 0, 0, (long)T9 * DFFC, 1);
      gemm(h16, d_w2t, dec_b2, tmp3, nullptr, T3, DD, DFFC, DFFC, 2048, DD, 0,
           (long)T9 * DFFC, 0, (long)T3 * DD, 0, 0);
      lnk(xw, tmp3, dec_ln3g, dec_ln3b, nullptr, xw, x16, T3,
          (long)T3 * DD, (long)T3 * DD, (long)T3 * DD, (long)T3 * DP);
    }

    float* out_r = (float*)d_out + (size_t)r * BB * NN * 2;
    k_offset<<<dim3((BB * NN + 255) / 256), 256, 0, stream>>>(xw, off_w, off_b, pre_bou, out_r, cur_b);
  }
}
